// GAT_Comm_80771154969225
// MI455X (gfx1250) — matrix operand order repaired and checked by reading, not compiled, not run on hardware
//
#include <hip/hip_runtime.h>
#include <hip/hip_bf16.h>
#include <math.h>

// ---------------- problem constants (match reference) ----------------
#define B_     32
#define N_     512
#define IN_    128
#define HID_   32
#define HEADS_ 4
#define OUT_   128
#define TOK_   (B_ * N_)          // 16384 tokens

typedef __attribute__((ext_vector_type(16))) _Float16 v16h;
typedef __attribute__((ext_vector_type(8)))  float    v8f;

union Frag16 {
    v16h v;
    _Float16 h[16];
    unsigned u[8];
};

// ---------------- wave32 reductions ----------------
__device__ __forceinline__ float wave_sum32(float v) {
    #pragma unroll
    for (int off = 16; off > 0; off >>= 1) v += __shfl_down(v, off, 32);
    return __shfl(v, 0, 32);
}

__device__ __forceinline__ float act_apply(float v, int ACT) {
    if (ACT == 1) v = v > 0.f ? v : expm1f(v);                       // ELU
    if (ACT == 2) v = 0.5f * v * (1.f + erff(v * 0.70710678118f));   // exact GELU
    return v;
}

// =====================================================================
// Strip-GEMM: C[M,128] = act(A[M,K] @ W[K,128] + bias)
// One wave computes a full 16x128 output strip: the A fragment is
// loaded ONCE per 32-wide k-step and reused across all 8 column tiles
// (8x reduction in A traffic), with 8 fp32 WMMA accumulators resident.
// Bias/activation resolved at compile time -> straight-line epilogue.
// =====================================================================
template<int ACT, bool HAS_BIAS>
__global__ __launch_bounds__(32)
void gemm_strip_wmma(const float* __restrict__ A, const float* __restrict__ W,
                     const float* __restrict__ bias, float* __restrict__ C,
                     int K)
{
    constexpr int NOUT = 128;
    constexpr int NT   = NOUT / 16;      // 8 column tiles
    const int lane = threadIdx.x & 31;
    const int half = lane >> 4;          // 0 or 1
    const int l16  = lane & 15;
    const int m0   = blockIdx.x << 4;
    const int mrow = m0 + l16;           // A row this lane supplies

    v8f acc[NT];
    #pragma unroll
    for (int t = 0; t < NT; ++t)
        acc[t] = (v8f){0.f, 0.f, 0.f, 0.f, 0.f, 0.f, 0.f, 0.f};

    for (int kb = 0; kb < K; kb += 32) {
        // --- A fragment, loaded once, reused for 8 WMMAs ---
        Frag16 af;
        #pragma unroll
        for (int r = 0; r < 8; ++r) {
            const int grp = r >> 2, rr = r & 3;
            const int ka  = kb + grp * 16 + half * 8 + rr * 2;
            const float* ap = A + (size_t)mrow * K + ka;
            af.h[2 * r]     = (_Float16)ap[0];
            af.h[2 * r + 1] = (_Float16)ap[1];
        }
        #pragma unroll
        for (int t = 0; t < NT; ++t) {
            const int ncol = t * 16 + l16;
            Frag16 bf;
            #pragma unroll
            for (int r = 0; r < 8; ++r) {
                const int kw = kb + (r >> 2) * 16 + half * 8 + (r & 3) * 2;
                bf.h[2 * r]     = (_Float16)W[(size_t)kw       * NOUT + ncol];
                bf.h[2 * r + 1] = (_Float16)W[(size_t)(kw + 1) * NOUT + ncol];
            }
            acc[t] = __builtin_amdgcn_wmma_f32_16x16x32_f16(
                         false, af.v, false, bf.v, (short)0, acc[t], false, false);
        }
    }

    #pragma unroll
    for (int t = 0; t < NT; ++t) {
        const int ncol = t * 16 + l16;
        const float bv = HAS_BIAS ? bias[ncol] : 0.f;
        #pragma unroll
        for (int r = 0; r < 8; ++r) {
            const int row = m0 + r + half * 8;     // C layout: M = r + 8*half
            float v = acc[t][r] + bv;
            v = act_apply(v, ACT);
            C[(size_t)row * NOUT + ncol] = v;
        }
    }
}

// =====================================================================
// Attention logits: e_src/e_dst = h . a  (tiny D-dot per (b,head,n))
// =====================================================================
template<int H, int D>
__global__ void compute_e_kernel(const float* __restrict__ hmat,
                                 const float* __restrict__ a_src,
                                 const float* __restrict__ a_dst,
                                 float* __restrict__ esrc,
                                 float* __restrict__ edst)
{
    const int idx = blockIdx.x * blockDim.x + threadIdx.x;   // (b*H+head)*N + n
    if (idx >= B_ * H * N_) return;
    const int n    = idx % N_;
    const int bh   = idx / N_;
    const int head = bh % H;
    const int b    = bh / H;
    float s = 0.f, d0 = 0.f;
    const float* hp = hmat + ((size_t)(b * N_ + n) * H + head) * D;
    #pragma unroll 4
    for (int d = 0; d < D; ++d) {
        const float v = hp[d];
        s  += v * a_src[head * D + d];
        d0 += v * a_dst[head * D + d];
    }
    esrc[idx] = s;
    edst[idx] = d0;
}

// =====================================================================
// Fused GAT attention: per (b, head, 16-row tile)
//   masked leaky-relu logits -> softmax (att stays in LDS as f16)
//   -> out[16,D] = att[16,512] @ h[512,D] via WMMA, scaled by 1/rowsum.
// Attention matrix NEVER touches HBM (saves ~270 MB of traffic).
// =====================================================================
template<int H, int D, int ACT>
__global__ __launch_bounds__(128)
void attn_wmma(const float* __restrict__ hmat, const float* __restrict__ esrc,
               const float* __restrict__ edst, const int* __restrict__ graph,
               float* __restrict__ out)
{
    __shared__ _Float16 att[16][N_ + 8];      // f16 softmax tile, padded
    __shared__ float red[16][8];
    __shared__ float rowmax[16];
    __shared__ float rowinv[16];

    const int b    = blockIdx.y / H;
    const int head = blockIdx.y % H;
    const int i0   = blockIdx.x << 4;
    const int tid  = threadIdx.x;

    // ---- phase 1: masked leaky-relu logits + 2-pass softmax into LDS ----
    {
        const int row = tid >> 3, sub = tid & 7;
        const int i   = i0 + row;
        const float es = esrc[((size_t)b * H + head) * N_ + i];
        const float* ed = edst + ((size_t)b * H + head) * N_;
        const int*   g  = graph + ((size_t)b * N_ + i) * N_;

        float lmax = -3.4e38f;
        for (int j = sub; j < N_; j += 8) {
            const float s  = es + ed[j];
            const float lr = s > 0.f ? s : 0.2f * s;
            const bool adj = (g[j] > 0) || (j == i);
            const float v  = adj ? lr : -30000.f;     // f16-safe "-inf"
            att[row][j] = (_Float16)v;
            lmax = fmaxf(lmax, v);
        }
        red[row][sub] = lmax;
        __syncthreads();
        if (sub == 0) {
            float m = red[row][0];
            #pragma unroll
            for (int t = 1; t < 8; ++t) m = fmaxf(m, red[row][t]);
            rowmax[row] = m;
        }
        __syncthreads();
        const float m = rowmax[row];
        float lsum = 0.f;
        for (int j = sub; j < N_; j += 8) {
            const float p = __expf((float)att[row][j] - m);
            att[row][j] = (_Float16)p;
            lsum += p;
        }
        red[row][sub] = lsum;
        __syncthreads();
        if (sub == 0) {
            float s2 = 0.f;
            #pragma unroll
            for (int t = 0; t < 8; ++t) s2 += red[row][t];
            rowinv[row] = 1.f / s2;
        }
        __syncthreads();
    }

    // ---- phase 2: out = att @ h, WMMA, column tiles distributed on waves ----
    const int wave = tid >> 5;
    const int lane = tid & 31;
    const int half = lane >> 4;
    const int l16  = lane & 15;

    for (int ct = wave; ct < D / 16; ct += 4) {
        const int col = ct * 16 + l16;
        v8f acc = {0.f, 0.f, 0.f, 0.f, 0.f, 0.f, 0.f, 0.f};

        for (int kt = 0; kt < N_ / 32; ++kt) {
            const int kb = kt * 32;
            Frag16 af, bf;
            #pragma unroll
            for (int r = 0; r < 8; ++r) {
                // A fragment straight from LDS (contiguous f16 K-pair = b32 read)
                const int grp = r >> 2, rr = r & 3;
                const int ka  = kb + grp * 16 + half * 8 + rr * 2;
                af.u[r] = *reinterpret_cast<const unsigned*>(&att[l16][ka]);
                // B fragment: h rows k, k+1 of column `col`
                const int kw = kb + grp * 16 + half * 8 + rr * 2;
                const size_t idx = ((size_t)(b * N_ + kw) * H + head) * D + col;
                bf.h[2 * r]     = (_Float16)hmat[idx];
                bf.h[2 * r + 1] = (_Float16)hmat[idx + (size_t)H * D];
            }
            acc = __builtin_amdgcn_wmma_f32_16x16x32_f16(
                      false, af.v, false, bf.v, (short)0, acc, false, false);
        }

        #pragma unroll
        for (int r = 0; r < 8; ++r) {
            const int mr = r + half * 8;
            float v = acc[r] * rowinv[mr];            // softmax normalization
            v = act_apply(v, ACT);
            out[(size_t)(b * N_ + (i0 + mr)) * (H * D) + head * D + col] = v;
        }
    }
}

// =====================================================================
// LayerNorm over concat([x(128), m(128)]) -> out[token,256]
// one wave32 per token, shuffle reductions
// =====================================================================
__global__ __launch_bounds__(32)
void ln_concat_kernel(const float* __restrict__ x, const float* __restrict__ m,
                      const float* __restrict__ g, const float* __restrict__ bb,
                      float* __restrict__ out)
{
    const int t    = blockIdx.x;
    const int lane = threadIdx.x;
    float vals[8];
    #pragma unroll
    for (int k = 0; k < 8; ++k) {
        const int f = lane + 32 * k;
        vals[k] = (f < 128) ? x[(size_t)t * 128 + f]
                            : m[(size_t)t * 128 + (f - 128)];
    }
    float s = 0.f;
    #pragma unroll
    for (int k = 0; k < 8; ++k) s += vals[k];
    const float mu = wave_sum32(s) * (1.f / 256.f);
    float vs = 0.f;
    #pragma unroll
    for (int k = 0; k < 8; ++k) { const float d = vals[k] - mu; vs += d * d; }
    const float rs = rsqrtf(wave_sum32(vs) * (1.f / 256.f) + 1e-5f);
    #pragma unroll
    for (int k = 0; k < 8; ++k) {
        const int f = lane + 32 * k;
        out[(size_t)t * 256 + f] = (vals[k] - mu) * rs * g[f] + bb[f];
    }
}

// LayerNorm over 128 features, residual presence compile-time
template<bool HAS_RES>
__global__ __launch_bounds__(32)
void ln128_kernel(const float* __restrict__ a, const float* __restrict__ res,
                  const float* __restrict__ g, const float* __restrict__ bb,
                  float* __restrict__ out)
{
    const int t    = blockIdx.x;
    const int lane = threadIdx.x;
    float vals[4];
    #pragma unroll
    for (int k = 0; k < 4; ++k) {
        const int f = lane + 32 * k;
        float v = a[(size_t)t * 128 + f];
        if (HAS_RES) v += res[(size_t)t * 128 + f];
        vals[k] = v;
    }
    float s = 0.f;
    #pragma unroll
    for (int k = 0; k < 4; ++k) s += vals[k];
    const float mu = wave_sum32(s) * (1.f / 128.f);
    float vs = 0.f;
    #pragma unroll
    for (int k = 0; k < 4; ++k) { const float d = vals[k] - mu; vs += d * d; }
    const float rs = rsqrtf(wave_sum32(vs) * (1.f / 128.f) + 1e-5f);
    #pragma unroll
    for (int k = 0; k < 4; ++k) {
        const int f = lane + 32 * k;
        out[(size_t)t * 128 + f] = (vals[k] - mu) * rs * g[f] + bb[f];
    }
}

// =====================================================================
// Host-side orchestration
// =====================================================================
extern "C" void kernel_launch(void* const* d_in, const int* in_sizes, int n_in,
                              void* d_out, int out_size, void* d_ws, size_t ws_size,
                              hipStream_t stream) {
    (void)in_sizes; (void)n_in; (void)out_size; (void)ws_size;

    const float* x      = (const float*)d_in[0];
    const int*   graph  = (const int*)  d_in[1];
    const float* W0     = (const float*)d_in[2];
    const float* a_src0 = (const float*)d_in[3];
    const float* a_dst0 = (const float*)d_in[4];
    const float* W1     = (const float*)d_in[5];
    const float* a_src1 = (const float*)d_in[6];
    const float* a_dst1 = (const float*)d_in[7];
    const float* ln1_g  = (const float*)d_in[8];
    const float* ln1_b  = (const float*)d_in[9];
    const float* lin_W  = (const float*)d_in[10];
    const float* lin_b  = (const float*)d_in[11];
    const float* me_W0  = (const float*)d_in[12];
    const float* me_b0  = (const float*)d_in[13];
    const float* me_W1  = (const float*)d_in[14];
    const float* me_b1  = (const float*)d_in[15];
    const float* ln2_g  = (const float*)d_in[16];
    const float* ln2_b  = (const float*)d_in[17];
    const float* oh_W   = (const float*)d_in[18];
    const float* oh_b   = (const float*)d_in[19];
    const float* ln3_g  = (const float*)d_in[20];
    const float* ln3_b  = (const float*)d_in[21];

    // workspace layout (float words)
    const size_t FEAT = (size_t)TOK_ * 128;      // 2M floats
    float* ws    = (float*)d_ws;
    float* h0    = ws;                 // [TOK,128]  GAT0 projection
    float* m1    = ws + 2 * FEAT;      // [TOK,128]  elu(GAT0 out)
    float* h1    = ws + 4 * FEAT;      // [TOK,128]  GAT1 projection
    float* m2    = ws + 6 * FEAT;      // [TOK,128]  GAT1 out (avg, H=1)
    float* ln1o  = ws + 8 * FEAT;      // [TOK,256]
    float* mfeat = ws + 12 * FEAT;     // [TOK,128]  linear out
    float* enc0g = ws + 14 * FEAT;     // [TOK,128]  gelu(me0)
    float* enc1  = ws + 16 * FEAT;     // [TOK,128]
    float* res2  = ws + 18 * FEAT;     // [TOK,128]  ln2 out
    float* ohg   = ws + 20 * FEAT;     // [TOK,128]  gelu(oh)
    float* es0   = ws + 22 * FEAT;                         // [B,4,N]
    float* ed0   = es0 + (size_t)B_ * HEADS_ * N_;
    float* es1   = ed0 + (size_t)B_ * HEADS_ * N_;         // [B,1,N]
    float* ed1   = es1 + (size_t)B_ * N_;

    const dim3 stripGrid(TOK_ / 16);   // 1024 waves, each owns a 16x128 strip
    const dim3 stripBlk(32);

    // ---- GAT layer 0 ----
    gemm_strip_wmma<0, false><<<stripGrid, stripBlk, 0, stream>>>(x, W0, nullptr, h0, 128);
    compute_e_kernel<HEADS_, HID_><<<(B_ * HEADS_ * N_ + 255) / 256, 256, 0, stream>>>(
        h0, a_src0, a_dst0, es0, ed0);
    attn_wmma<HEADS_, HID_, 1><<<dim3(N_ / 16, B_ * HEADS_), 128, 0, stream>>>(
        h0, es0, ed0, graph, m1);

    // ---- GAT layer 1 (H=1, D=128, mean over 1 head == identity) ----
    gemm_strip_wmma<0, false><<<stripGrid, stripBlk, 0, stream>>>(m1, W1, nullptr, h1, 128);
    compute_e_kernel<1, OUT_><<<(B_ * N_ + 255) / 256, 256, 0, stream>>>(
        h1, a_src1, a_dst1, es1, ed1);
    attn_wmma<1, OUT_, 0><<<dim3(N_ / 16, B_), 128, 0, stream>>>(
        h1, es1, ed1, graph, m2);

    // ---- tail: LN1(concat) -> linear -> encoder MLP -> LN2 -> head -> LN3 ----
    ln_concat_kernel<<<TOK_, 32, 0, stream>>>(x, m2, ln1_g, ln1_b, ln1o);
    gemm_strip_wmma<0, true><<<stripGrid, stripBlk, 0, stream>>>(ln1o, lin_W, lin_b, mfeat, 256);
    gemm_strip_wmma<2, true><<<stripGrid, stripBlk, 0, stream>>>(mfeat, me_W0, me_b0, enc0g, 128);
    gemm_strip_wmma<0, true><<<stripGrid, stripBlk, 0, stream>>>(enc0g, me_W1, me_b1, enc1, 128);
    ln128_kernel<true><<<TOK_, 32, 0, stream>>>(mfeat, enc1, ln2_g, ln2_b, res2);
    gemm_strip_wmma<2, true><<<stripGrid, stripBlk, 0, stream>>>(res2, oh_W, oh_b, ohg, 128);
    ln128_kernel<false><<<TOK_, 32, 0, stream>>>(ohg, nullptr, ln3_g, ln3_b, (float*)d_out);
}
